// WindowAttentionNew_33921651704467
// MI455X (gfx1250) — compile-verified, matrix operand order checked by reading, not run on hardware
//
#include <hip/hip_runtime.h>

#define H_    12
#define NTOK  49
#define DH    32
#define CDIM  384
#define BWIN  2048
#define NWIN  64
#define MTOT  (BWIN * NTOK)      // 100352
#define QKVN  (3 * CDIM)         // 1152

typedef __attribute__((ext_vector_type(16))) _Float16 v16h;
typedef __attribute__((ext_vector_type(8)))  float    v8f;
typedef __attribute__((ext_vector_type(2)))  __fp16   v2fp;

union F8  { v8f v; float f[8]; };
union H16 { v16h v; unsigned int u[8]; _Float16 h[16]; };

__device__ inline v8f vzero8() {
  v8f z = {0.f, 0.f, 0.f, 0.f, 0.f, 0.f, 0.f, 0.f};
  return z;
}

// Packed f32x4 -> f16x4 via v_cvt_pk_rtz_f16_f32 (2 converts instead of 4+pack).
__device__ inline unsigned long long cvt4(float4 v) {
  union { v2fp h2[2]; unsigned long long u; } p;
  p.h2[0] = __builtin_amdgcn_cvt_pkrtz(v.x, v.y);
  p.h2[1] = __builtin_amdgcn_cvt_pkrtz(v.z, v.w);
  return p.u;
}

// A-fragment (16x32 f16, M-major [row][k]); also the B-fragment when the
// logical B(k,n) comes from an N-major matrix [n][k].
// ISA: lanes 0-15: M=lane, VGPR0-3 K=0..7 pairs, VGPR4-7 K=16..23;
// lanes 16-31: M=lane-16, VGPR0-3 K=8..15, VGPR4-7 K=24..31.
__device__ inline v16h frag_nmajor(const _Float16* s, int r0, int k0, int ld, int lane) {
  int m = lane & 15, hf = lane >> 4;
  H16 a;
#pragma unroll
  for (int j = 0; j < 8; ++j) {
    int kb = ((j >> 2) << 4) + (hf << 3) + ((j & 3) << 1);
    a.u[j] = *reinterpret_cast<const unsigned int*>(s + (r0 + m) * ld + (k0 + kb));
  }
  return a.v;
}

// B-fragment (32x16 f16) from a K-major matrix [k][n] (V for P·V).
__device__ inline v16h frag_kmajor(const _Float16* s, int k0, int n0, int ld, int lane) {
  int n = lane & 15, hf = lane >> 4;
  H16 a;
#pragma unroll
  for (int j = 0; j < 8; ++j) {
    int kb = ((j >> 2) << 4) + (hf << 3) + ((j & 3) << 1);
    a.h[2 * j]     = s[(k0 + kb)     * ld + n0 + n];
    a.h[2 * j + 1] = s[(k0 + kb + 1) * ld + n0 + n];
  }
  return a.v;
}

__device__ inline v8f wmma16(v16h a, v16h b, v8f c) {
  return __builtin_amdgcn_wmma_f32_16x16x32_f16(false, a, false, b, (short)0, c,
                                                false, false);
}

// ---------------- Kernel 1: CPB MLP table + clamped logit scale -------------
__global__ void cpb_kernel(const float* __restrict__ rel_table,
                           const float* __restrict__ cpb_w1,
                           const float* __restrict__ cpb_b1,
                           const float* __restrict__ cpb_w2,
                           const float* __restrict__ logit_scale,
                           float* __restrict__ tbl16s, float* __restrict__ ls) {
  int idx = blockIdx.x * blockDim.x + threadIdx.x;
  if (idx < H_) {
    float v = logit_scale[idx];
    ls[idx] = __expf(fminf(v, 4.6051701859880914f));   // log(100)
  }
  if (idx >= 169 * H_) return;
  int pos = idx / H_, head = idx % H_;
  float t0 = rel_table[pos * 2], t1 = rel_table[pos * 2 + 1];
  float acc = 0.f;
  for (int i = 0; i < 512; ++i) {
    float hv = fmaf(cpb_w1[2 * i], t0, fmaf(cpb_w1[2 * i + 1], t1, cpb_b1[i]));
    hv = fmaxf(hv, 0.f);
    acc = fmaf(cpb_w2[head * 512 + i], hv, acc);
  }
  tbl16s[idx] = 16.f / (1.f + __expf(-acc));           // 16*sigmoid
}

// ---------------- Kernel 2: combined bias = rpb + shift mask ----------------
__global__ void bias_kernel(const float* __restrict__ tbl16s,
                            const int* __restrict__ rel_index,
                            const float* __restrict__ mask,
                            float* __restrict__ biasbuf) {
  int idx = blockIdx.x * blockDim.x + threadIdx.x;
  const int NN = NTOK * NTOK;
  if (idx >= NWIN * H_ * NN) return;
  int nm = idx % NN;
  int wh = idx / NN;
  int h = wh % H_, w = wh / H_;
  biasbuf[idx] = tbl16s[rel_index[nm] * H_ + h] + mask[w * NN + nm];
}

// ---------------- Kernel 3: QKV GEMM (f32 in, f16 WMMA, f16 scatter) --------
#define BM  64
#define BN  128
#define BK  32
#define KT  (CDIM / BK)   // 12
#define LDT 40

__global__ void __launch_bounds__(128)
qkv_gemm(const float* __restrict__ x, const float* __restrict__ qkv_w,
         const float* __restrict__ q_bias, const float* __restrict__ v_bias,
         _Float16* __restrict__ qh, _Float16* __restrict__ kh,
         _Float16* __restrict__ vh) {
  __shared__ __align__(16) _Float16 sA[BM * LDT];
  __shared__ __align__(16) _Float16 sB[BN * LDT];
  int tid = threadIdx.x, lane = tid & 31, wv = tid >> 5;
  int m0 = blockIdx.x * BM;
  int n0 = blockIdx.y * BN;
  v8f acc[8];
#pragma unroll
  for (int i = 0; i < 8; ++i) acc[i] = vzero8();

  // Tile staging: 8 threads/row, float4 per thread, 16 rows per sweep.
  int rT = tid >> 3, cT = (tid & 7) * 4;
  const float* gA = x + (size_t)(m0 + rT) * CDIM + cT;
  const float* gB = qkv_w + (size_t)(n0 + rT) * CDIM + cT;
  float4 fa4[4], fb4[8];

  auto fetch = [&](int kk) {
#pragma unroll
    for (int s = 0; s < 4; ++s)
      fa4[s] = *reinterpret_cast<const float4*>(gA + (size_t)s * 16 * CDIM + kk);
#pragma unroll
    for (int s = 0; s < 8; ++s)
      fb4[s] = *reinterpret_cast<const float4*>(gB + (size_t)s * 16 * CDIM + kk);
  };
  auto stash = [&]() {
#pragma unroll
    for (int s = 0; s < 4; ++s)
      *reinterpret_cast<unsigned long long*>(sA + (rT + s * 16) * LDT + cT) = cvt4(fa4[s]);
#pragma unroll
    for (int s = 0; s < 8; ++s)
      *reinterpret_cast<unsigned long long*>(sB + (rT + s * 16) * LDT + cT) = cvt4(fb4[s]);
  };

  fetch(0);
  for (int kt = 0; kt < KT; ++kt) {
    stash();
    __syncthreads();
    if (kt + 1 < KT) fetch((kt + 1) * BK);   // overlap next loads with WMMA
    v16h a = frag_nmajor(sA, 16 * wv, 0, LDT, lane);
    // B-fragment loads rotated one step ahead of the consuming WMMA.
    v16h bcur = frag_nmajor(sB, 0, 0, LDT, lane);
#pragma unroll
    for (int nt = 0; nt < 8; ++nt) {
      v16h bnxt = (nt < 7) ? frag_nmajor(sB, 16 * (nt + 1), 0, LDT, lane) : bcur;
      acc[nt] = wmma16(a, bcur, acc[nt]);
      bcur = bnxt;
    }
    __syncthreads();
  }

  int cn = lane & 15, hf = lane >> 4;
#pragma unroll
  for (int r = 0; r < 8; ++r) {
    int m = m0 + 16 * wv + r + 8 * hf;
    int bwin = m / NTOK, ntok = m % NTOK;
#pragma unroll
    for (int nt = 0; nt < 8; ++nt) {
      int j = n0 + 16 * nt + cn;
      int sel = j / CDIM, c = j % CDIM;
      float bias = (sel == 0) ? q_bias[c] : ((sel == 2) ? v_bias[c] : 0.f);
      F8 fa; fa.v = acc[nt];
      float val = fa.f[r] + bias;
      int head = c >> 5, d = c & 31;
      size_t dst = ((size_t)(bwin * H_ + head) * NTOK + ntok) * DH + d;
      _Float16 hv = (_Float16)val;
      if (sel == 0)      qh[dst] = hv;
      else if (sel == 1) kh[dst] = hv;
      else               vh[dst] = hv;
    }
  }
}

// ---------------- Kernel 4: fused cosine attention per (window, head) -------
#define QLD 40
#define PLD 72

__global__ void __launch_bounds__(128)
attn_kernel(const _Float16* __restrict__ qh, const _Float16* __restrict__ kh,
            const _Float16* __restrict__ vh, const float* __restrict__ biasbuf,
            const float* __restrict__ ls, _Float16* __restrict__ ao) {
  __shared__ __align__(16) _Float16 sQ[64 * QLD];
  __shared__ __align__(16) _Float16 sK[64 * QLD];
  __shared__ __align__(16) _Float16 sV[64 * QLD];
  __shared__ __align__(16) _Float16 sP[64 * PLD];

  int tid = threadIdx.x, lane = tid & 31, wv = tid >> 5;
  int bh = blockIdx.x;                       // b*H + h
  int b = bh / H_, h = bh % H_;
  int w = b & (NWIN - 1);
  size_t base = (size_t)bh * NTOK * DH;

  // Stage q/k/v tiles vectorized (8 halves = 16B per thread), zero-pad 49..63.
  {
    int rs = tid >> 2, cs = (tid & 3) * 8;
    float4 zf = make_float4(0.f, 0.f, 0.f, 0.f);
#pragma unroll
    for (int s = 0; s < 2; ++s) {
      int r = rs + s * 32;
      bool ok = r < NTOK;
      size_t goff = base + (size_t)r * DH + cs;
      float4 vq = ok ? *reinterpret_cast<const float4*>(qh + goff) : zf;
      float4 vk = ok ? *reinterpret_cast<const float4*>(kh + goff) : zf;
      float4 vs = ok ? *reinterpret_cast<const float4*>(vh + goff) : zf;
      *reinterpret_cast<float4*>(sQ + r * QLD + cs) = vq;
      *reinterpret_cast<float4*>(sK + r * QLD + cs) = vk;
      *reinterpret_cast<float4*>(sV + r * QLD + cs) = vs;
    }
  }
  __syncthreads();

  // Cosine norm; fold clamped logit scale into q.
  {
    int r = tid & 63;
    _Float16* sm = (tid < 64) ? sQ : sK;
    float extra = (tid < 64) ? ls[h] : 1.f;
    float ss = 0.f;
#pragma unroll
    for (int c = 0; c < 32; ++c) { float t = (float)sm[r * QLD + c]; ss = fmaf(t, t, ss); }
    float sc = extra / (sqrtf(ss) + 1e-6f);
#pragma unroll
    for (int c = 0; c < 32; ++c)
      sm[r * QLD + c] = (_Float16)((float)sm[r * QLD + c] * sc);
  }
  __syncthreads();

  // S = qn @ kn^T  (K = Dh = 32 -> one WMMA per 16x16 tile)
  v8f s4[4];
  v16h aq = frag_nmajor(sQ, 16 * wv, 0, QLD, lane);
  {
    v16h bcur = frag_nmajor(sK, 0, 0, QLD, lane);
#pragma unroll
    for (int nt = 0; nt < 4; ++nt) {
      v16h bnxt = (nt < 3) ? frag_nmajor(sK, 16 * (nt + 1), 0, QLD, lane) : bcur;
      s4[nt] = wmma16(aq, bcur, vzero8());
      bcur = bnxt;
    }
  }

  int cn = lane & 15, hf = lane >> 4;
  const float* bb = biasbuf + (size_t)(w * H_ + h) * (NTOK * NTOK);
  float sv[4][8];
#pragma unroll
  for (int nt = 0; nt < 4; ++nt) {
    F8 fa; fa.v = s4[nt];
#pragma unroll
    for (int r = 0; r < 8; ++r) {
      int row = 16 * wv + r + 8 * hf;
      int col = 16 * nt + cn;
      float s;
      if (col >= NTOK)      s = -1e30f;                 // padded keys
      else if (row < NTOK)  s = fa.f[r] + bb[row * NTOK + col];
      else                  s = fa.f[r];                // padded queries: don't care
      sv[nt][r] = s;
    }
  }

  // Softmax: each row lives in one 16-lane half -> butterfly masks 1,2,4,8.
#pragma unroll
  for (int r = 0; r < 8; ++r) {
    float mx = sv[0][r];
#pragma unroll
    for (int nt = 1; nt < 4; ++nt) mx = fmaxf(mx, sv[nt][r]);
    for (int off = 1; off < 16; off <<= 1) mx = fmaxf(mx, __shfl_xor(mx, off, 32));
    float sum = 0.f;
#pragma unroll
    for (int nt = 0; nt < 4; ++nt) {
      float e = __expf(sv[nt][r] - mx);
      sv[nt][r] = e;
      sum += e;
    }
    for (int off = 1; off < 16; off <<= 1) sum += __shfl_xor(sum, off, 32);
    float inv = 1.f / (sum + 1e-6f);
    int row = 16 * wv + r + 8 * hf;
#pragma unroll
    for (int nt = 0; nt < 4; ++nt)
      sP[row * PLD + 16 * nt + cn] = (_Float16)(sv[nt][r] * inv);
  }
  __syncthreads();

  // O = P @ V  (K = 64 tokens -> 2 WMMA K-steps, N = 32)
  v8f o2[2];
  o2[0] = vzero8(); o2[1] = vzero8();
#pragma unroll
  for (int kk = 0; kk < 64; kk += 32) {
    v16h ap = frag_nmajor(sP, 16 * wv, kk, PLD, lane);
#pragma unroll
    for (int nt = 0; nt < 2; ++nt) {
      v16h bv = frag_kmajor(sV, kk, 16 * nt, QLD, lane);
      o2[nt] = wmma16(ap, bv, o2[nt]);
    }
  }
#pragma unroll
  for (int nt = 0; nt < 2; ++nt) {
    F8 fo; fo.v = o2[nt];
#pragma unroll
    for (int r = 0; r < 8; ++r) {
      int row = 16 * wv + r + 8 * hf;
      if (row < NTOK) {
        int d = 16 * nt + cn;
        ao[((size_t)b * NTOK + row) * CDIM + h * DH + d] = (_Float16)fo.f[r];
      }
    }
  }
}

// ---------------- Kernel 5: output projection GEMM (f16 -> f32 + bias) ------
__global__ void __launch_bounds__(128)
proj_gemm(const _Float16* __restrict__ ao, const float* __restrict__ proj_w,
          const float* __restrict__ proj_b, float* __restrict__ out) {
  __shared__ __align__(16) _Float16 sA[BM * LDT];
  __shared__ __align__(16) _Float16 sB[BN * LDT];
  int tid = threadIdx.x, lane = tid & 31, wv = tid >> 5;
  int m0 = blockIdx.x * BM;
  int n0 = blockIdx.y * BN;
  v8f acc[8];
#pragma unroll
  for (int i = 0; i < 8; ++i) acc[i] = vzero8();

  // A tile: already f16 -> raw 16B copies (4 threads/row, 32 rows/sweep).
  int rA = tid >> 2, cA = (tid & 3) * 8;
  const _Float16* gA = ao + (size_t)(m0 + rA) * CDIM + cA;
  // B tile: f32 weights -> convert (8 threads/row, 16 rows/sweep).
  int rB = tid >> 3, cB = (tid & 7) * 4;
  const float* gB = proj_w + (size_t)(n0 + rB) * CDIM + cB;
  float4 ha4[2], fb4[8];

  auto fetch = [&](int kk) {
#pragma unroll
    for (int s = 0; s < 2; ++s)
      ha4[s] = *reinterpret_cast<const float4*>(gA + (size_t)s * 32 * CDIM + kk);
#pragma unroll
    for (int s = 0; s < 8; ++s)
      fb4[s] = *reinterpret_cast<const float4*>(gB + (size_t)s * 16 * CDIM + kk);
  };
  auto stash = [&]() {
#pragma unroll
    for (int s = 0; s < 2; ++s)
      *reinterpret_cast<float4*>(sA + (rA + s * 32) * LDT + cA) = ha4[s];
#pragma unroll
    for (int s = 0; s < 8; ++s)
      *reinterpret_cast<unsigned long long*>(sB + (rB + s * 16) * LDT + cB) = cvt4(fb4[s]);
  };

  fetch(0);
  for (int kt = 0; kt < KT; ++kt) {
    stash();
    __syncthreads();
    if (kt + 1 < KT) fetch((kt + 1) * BK);
    v16h a = frag_nmajor(sA, 16 * wv, 0, LDT, lane);
    v16h bcur = frag_nmajor(sB, 0, 0, LDT, lane);
#pragma unroll
    for (int nt = 0; nt < 8; ++nt) {
      v16h bnxt = (nt < 7) ? frag_nmajor(sB, 16 * (nt + 1), 0, LDT, lane) : bcur;
      acc[nt] = wmma16(a, bcur, acc[nt]);
      bcur = bnxt;
    }
    __syncthreads();
  }

  int cn = lane & 15, hf = lane >> 4;
#pragma unroll
  for (int r = 0; r < 8; ++r) {
    int m = m0 + 16 * wv + r + 8 * hf;
#pragma unroll
    for (int nt = 0; nt < 8; ++nt) {
      int j = n0 + 16 * nt + cn;
      F8 fa; fa.v = acc[nt];
      out[(size_t)m * CDIM + j] = fa.f[r] + proj_b[j];
    }
  }
}

// ---------------------------------------------------------------------------
extern "C" void kernel_launch(void* const* d_in, const int* in_sizes, int n_in,
                              void* d_out, int out_size, void* d_ws, size_t ws_size,
                              hipStream_t stream) {
  (void)in_sizes; (void)n_in; (void)out_size; (void)ws_size;
  const float* x           = (const float*)d_in[0];
  const float* mask        = (const float*)d_in[1];
  const float* qkv_w       = (const float*)d_in[2];
  const float* q_bias      = (const float*)d_in[3];
  const float* v_bias      = (const float*)d_in[4];
  const float* logit_scale = (const float*)d_in[5];
  const float* cpb_w1      = (const float*)d_in[6];
  const float* cpb_b1      = (const float*)d_in[7];
  const float* cpb_w2      = (const float*)d_in[8];
  const float* proj_w      = (const float*)d_in[9];
  const float* proj_b      = (const float*)d_in[10];
  const float* rel_table   = (const float*)d_in[11];
  const int*   rel_index   = (const int*)d_in[12];

  char* ws = (char*)d_ws;
  float* ls      = (float*)(ws + 0);       // 12 f32
  float* tbl16s  = (float*)(ws + 64);      // 169*12 f32
  float* biasbuf = (float*)(ws + 8192);    // 64*12*49*49 f32 (~7.4 MB)
  size_t off = 8192 + (size_t)NWIN * H_ * NTOK * NTOK * sizeof(float);
  off = (off + 255) & ~(size_t)255;
  const size_t QKV_ELEMS = (size_t)BWIN * H_ * NTOK * DH;   // 38.5M f16 each
  _Float16* qh = (_Float16*)(ws + off); off += QKV_ELEMS * 2;
  _Float16* kh = (_Float16*)(ws + off); off += QKV_ELEMS * 2;
  _Float16* vh = (_Float16*)(ws + off); off += QKV_ELEMS * 2;
  _Float16* ao = (_Float16*)(ws + off);                     // MTOT*CDIM f16

  cpb_kernel<<<(169 * H_ + 255) / 256, 256, 0, stream>>>(
      rel_table, cpb_w1, cpb_b1, cpb_w2, logit_scale, tbl16s, ls);

  int btot = NWIN * H_ * NTOK * NTOK;
  bias_kernel<<<(btot + 255) / 256, 256, 0, stream>>>(tbl16s, rel_index, mask,
                                                      biasbuf);

  qkv_gemm<<<dim3(MTOT / BM, QKVN / BN), 128, 0, stream>>>(
      x, qkv_w, q_bias, v_bias, qh, kh, vh);

  attn_kernel<<<BWIN * H_, 128, 0, stream>>>(qh, kh, vh, biasbuf, ls, ao);

  proj_gemm<<<dim3(MTOT / BM, CDIM / BN), 128, 0, stream>>>(ao, proj_w, proj_b,
                                                            (float*)d_out);
}
